// MultRNN_64587718197506
// MI455X (gfx1250) — matrix operand order repaired and checked by reading, not compiled, not run on hardware
//
#include <hip/hip_runtime.h>

// ---------------------------------------------------------------------------
// MultRNN step on gfx1250:
//   - one-off f32->bf16 packing of all GEMM operands (v_cvt_pk_bf16_f32)
//   - three bf16 WMMA GEMMs whose tiles are staged with
//     GLOBAL_LOAD_ASYNC_TO_LDS_B128 (ASYNCcnt) -- zero staging VALU/ds_store
//   - D = A(bf16) x B(bf16)^T, f32 accumulation via v_wmma_f32_16x16x32_bf16
// ---------------------------------------------------------------------------

typedef __attribute__((ext_vector_type(16))) __bf16 v16bf;
typedef __attribute__((ext_vector_type(2)))  __bf16 v2bf;
typedef __attribute__((ext_vector_type(8)))  float  v8f;

#define BATCH 8192
#define IN_SZ 512
#define H_SZ  1024
#define F_SZ  512
#define O_SZ  1024

#define TM 128
#define TN 128
#define TK 32
#define LDSP 40  // padded LDS row stride in ushorts (80B -> conflict-free b128)

__device__ __forceinline__ unsigned short f2bfu(float x) {
  __bf16 h = (__bf16)x;
  unsigned short u;
  __builtin_memcpy(&u, &h, 2);
  return u;
}
__device__ __forceinline__ unsigned pk2(float a, float b) {
  v2bf t;
  t.x = (__bf16)a;
  t.y = (__bf16)b;
  unsigned u;
  __builtin_memcpy(&u, &t, 4);
  return u;
}

// Async DMA: global (saddr + 32-bit voffset) -> LDS, 16 bytes per lane.
__device__ __forceinline__ void async_b128(unsigned lds_off, unsigned byte_off,
                                           const void* base) {
  asm volatile("global_load_async_to_lds_b128 %0, %1, %2"
               :: "v"(lds_off), "v"(byte_off), "s"(base)
               : "memory");
}
__device__ __forceinline__ void wait_async0() {
  asm volatile("s_wait_asynccnt 0x0" ::: "memory");
}

union Frag { uint4 q[2]; v16bf v; };

// MODE: 0 = Cb16[r,c] = bf16(aux[c] * acc)        (f_t, bf16 only)
//       1 = Cf[r,c] = tanhf(acc); Cb2 = bf16(Cf)  (h_t f32 out + bf16 ws)
//       2 = Cf[r,c] = acc + aux[c]                (o_t)
// All A/B operands are bf16 [rows x K] row-major; A and B may be K-split.
template <int MODE>
__global__ __launch_bounds__(256) void gemm_wmma(
    void* __restrict__ Cv, int ldc, unsigned short* __restrict__ Cb2,
    const unsigned short* __restrict__ A1, const unsigned short* __restrict__ A2,
    int lda, int splitA,
    const unsigned short* __restrict__ B1, const unsigned short* __restrict__ B2,
    int ldb, int splitB,
    const float* __restrict__ aux, int K) {
  __shared__ unsigned short lA[2][TM][LDSP];
  __shared__ unsigned short lB[2][TN][LDSP];

  const int tid  = threadIdx.x;
  const int lane = tid & 31;
  const int w    = tid >> 5;   // wave 0..7
  const int wr   = w >> 2;     // 0..1 -> 64 rows each
  const int wc   = w & 3;      // 0..3 -> 32 cols each
  const int lrow = lane & 15;
  const int lh   = lane >> 4;

  const int r0 = blockIdx.x * TM;
  const int n0 = blockIdx.y * TN;

  v8f acc[4][2];
#pragma unroll
  for (int mt = 0; mt < 4; ++mt)
#pragma unroll
    for (int nt = 0; nt < 2; ++nt)
#pragma unroll
      for (int e = 0; e < 8; ++e) acc[mt][nt][e] = 0.0f;

  // Issue this thread's 4 async 16B chunks (2 for A, 2 for B) for one k-step.
  // Tile = 128 rows x 32 bf16 (64B/row = 4 chunks); 512 chunks / 256 threads.
  auto stage = [&](int buf, int k0) {
    const unsigned short* Ap;
    const unsigned short* Bp;
    int ka, kb;
    if (k0 < splitA) { Ap = A1; ka = k0; } else { Ap = A2; ka = k0 - splitA; }
    if (k0 < splitB) { Bp = B1; kb = k0; } else { Bp = B2; kb = k0 - splitB; }
#pragma unroll
    for (int i = 0; i < 2; ++i) {
      const int c = i * 256 + tid;
      const int row = c >> 2, q = c & 3;
      const unsigned aoff = (unsigned)(((size_t)(r0 + row) * lda + ka + (q << 3)) * 2);
      const unsigned boff = (unsigned)(((size_t)(n0 + row) * ldb + kb + (q << 3)) * 2);
      async_b128((unsigned)(size_t)&lA[buf][row][q << 3], aoff, Ap);
      async_b128((unsigned)(size_t)&lB[buf][row][q << 3], boff, Bp);
    }
  };

  stage(0, 0);
  wait_async0();
  __syncthreads();

  const int nK = K / TK;
  for (int kt = 0; kt < nK; ++kt) {
    const int cur = kt & 1;
    const bool more = (kt + 1) < nK;
    if (more) stage(cur ^ 1, (kt + 1) * TK);  // DMA overlaps the WMMAs below

    // Fragment loads per ISA VGPR layouts (16-bit A 16x32 / B 32x16).
    Frag a[4], bf[2];
#pragma unroll
    for (int mt = 0; mt < 4; ++mt) {
      const unsigned short* p = &lA[cur][wr * 64 + mt * 16 + lrow][0];
      a[mt].q[0] = *(const uint4*)(p + lh * 8);        // K = h*8 .. h*8+7
      a[mt].q[1] = *(const uint4*)(p + 16 + lh * 8);   // K = 16+h*8 ..
    }
#pragma unroll
    for (int nt = 0; nt < 2; ++nt) {
      const unsigned short* p = &lB[cur][wc * 32 + nt * 16 + lrow][0];
      bf[nt].q[0] = *(const uint4*)(p + lh * 8);       // K = h*8 .. h*8+7
      bf[nt].q[1] = *(const uint4*)(p + 16 + lh * 8);  // K = 16+h*8 ..
    }
#pragma unroll
    for (int mt = 0; mt < 4; ++mt)
#pragma unroll
      for (int nt = 0; nt < 2; ++nt)
        acc[mt][nt] = __builtin_amdgcn_wmma_f32_16x16x32_bf16(
            false, a[mt].v, false, bf[nt].v, (short)0, acc[mt][nt], false, false);

    if (more) wait_async0();  // this wave's nxt-buffer DMAs done
    __syncthreads();          // everyone's DMAs done / cur reads finished
  }

  // --- epilogue: C/D 16x16 f32 layout: VGPR r -> M = r + lh*8, N = lrow ----
#pragma unroll
  for (int mt = 0; mt < 4; ++mt)
#pragma unroll
    for (int nt = 0; nt < 2; ++nt) {
      const int gcol = n0 + wc * 32 + nt * 16 + lrow;
      const float cs = (MODE == 1) ? 0.0f : aux[gcol];
#pragma unroll
      for (int r = 0; r < 8; ++r) {
        const int grow = r0 + wr * 64 + mt * 16 + lh * 8 + r;
        const size_t idx = (size_t)grow * ldc + gcol;
        const float v = acc[mt][nt][r];
        if (MODE == 0) {
          ((unsigned short*)Cv)[idx] = f2bfu(cs * v);
        } else if (MODE == 1) {
          const float t = tanhf(v);
          ((float*)Cv)[idx] = t;
          Cb2[idx] = f2bfu(t);
        } else {
          ((float*)Cv)[idx] = v + cs;
        }
      }
    }
}

// left[f] = dot(x_t[f,:], W_fx[f,:]),  f < 512, K = 512
__global__ __launch_bounds__(256) void left_dot_kernel(
    const float* __restrict__ x, const float* __restrict__ Wfx,
    float* __restrict__ left) {
  const int f = blockIdx.x;
  float s = 0.0f;
  for (int k = threadIdx.x; k < IN_SZ; k += 256)
    s += x[f * IN_SZ + k] * Wfx[f * IN_SZ + k];
#pragma unroll
  for (int off = 16; off > 0; off >>= 1) s += __shfl_xor(s, off, 32);
  __shared__ float red[8];
  if ((threadIdx.x & 31) == 0) red[threadIdx.x >> 5] = s;
  __syncthreads();
  if (threadIdx.x == 0) {
    float t = 0.0f;
#pragma unroll
    for (int i = 0; i < 8; ++i) t += red[i];
    left[f] = t;
  }
}

// dst[i] = bf16(src[i]); one float4 -> uint2 per thread (exact grid)
__global__ __launch_bounds__(256) void cvt_bf16_kernel(
    const float* __restrict__ src, unsigned short* __restrict__ dst) {
  const size_t i = ((size_t)blockIdx.x * 256 + threadIdx.x) * 4;
  const float4 v = *(const float4*)(src + i);
  *(uint2*)(dst + i) = make_uint2(pk2(v.x, v.y), pk2(v.z, v.w));
}

// dst[n*rows + k] = bf16(src[k*cols + n])  -- 32x32 LDS-tiled transpose+convert
__global__ __launch_bounds__(256) void transpose_to_bf16(
    const float* __restrict__ src, unsigned short* __restrict__ dst,
    int rows /*K*/, int cols /*N*/) {
  __shared__ float tile[32][33];
  const int bn = blockIdx.x * 32;
  const int bk = blockIdx.y * 32;
  const int tx = threadIdx.x & 31;
  const int ty = threadIdx.x >> 5;
#pragma unroll
  for (int j = ty; j < 32; j += 8)
    tile[j][tx] = src[(size_t)(bk + j) * cols + bn + tx];
  __syncthreads();
#pragma unroll
  for (int j = ty; j < 32; j += 8)
    dst[(size_t)(bn + j) * rows + bk + tx] = f2bfu(tile[tx][j]);
}

extern "C" void kernel_launch(void* const* d_in, const int* in_sizes, int n_in,
                              void* d_out, int out_size, void* d_ws, size_t ws_size,
                              hipStream_t stream) {
  (void)in_sizes; (void)n_in; (void)out_size; (void)ws_size;
  const float* x_t   = (const float*)d_in[0];  // [8192, 512]
  const float* h_tm1 = (const float*)d_in[1];  // [8192, 1024]
  const float* W_fx  = (const float*)d_in[2];  // [512, 512]
  const float* W_fh  = (const float*)d_in[3];  // [512, 1024]
  const float* W_hf  = (const float*)d_in[4];  // [1024, 512]
  const float* W_hx  = (const float*)d_in[5];  // [1024, 512]
  const float* W_oh  = (const float*)d_in[6];  // [1024, 1024]
  const float* b_o   = (const float*)d_in[7];  // [1024]

  float* o_out = (float*)d_out;                         // [8192, 1024]
  float* h_out = (float*)d_out + (size_t)BATCH * H_SZ;  // [8192, 1024]

  // Workspace layout (16B aligned, ~37 MB):
  char* ws = (char*)d_ws;
  float*          left = (float*)ws;                       // 2 KB (pad 4 KB)
  unsigned short* hb   = (unsigned short*)(ws + 4096);     // h_tm1 bf16, 16 MB
  unsigned short* hb2  = hb;                               // h_t bf16 (aliases hb; hb dead after GEMM1)
  unsigned short* xb   = (unsigned short*)(ws + 4096 + (16u << 20));          // 8 MB
  unsigned short* ftb  = (unsigned short*)(ws + 4096 + (24u << 20));          // 8 MB
  unsigned short* Wfhb = (unsigned short*)(ws + 4096 + (32u << 20));          // 1 MB
  unsigned short* Whfb = (unsigned short*)(ws + 4096 + (33u << 20));          // 1 MB
  unsigned short* Whxb = (unsigned short*)(ws + 4096 + (34u << 20));          // 1 MB
  unsigned short* WohT = (unsigned short*)(ws + 4096 + (35u << 20));          // 2 MB

  // 0) tiny: left[f] = diag(x @ W_fx^T)
  left_dot_kernel<<<F_SZ, 256, 0, stream>>>(x_t, W_fx, left);

  // 1) pack operands to bf16
  cvt_bf16_kernel<<<(BATCH * H_SZ) / 1024, 256, 0, stream>>>(h_tm1, hb);
  cvt_bf16_kernel<<<(BATCH * IN_SZ) / 1024, 256, 0, stream>>>(x_t, xb);
  cvt_bf16_kernel<<<(F_SZ * H_SZ) / 1024, 256, 0, stream>>>(W_fh, Wfhb);
  cvt_bf16_kernel<<<(H_SZ * F_SZ) / 1024, 256, 0, stream>>>(W_hf, Whfb);
  cvt_bf16_kernel<<<(H_SZ * IN_SZ) / 1024, 256, 0, stream>>>(W_hx, Whxb);
  transpose_to_bf16<<<dim3(O_SZ / 32, H_SZ / 32), 256, 0, stream>>>(
      W_oh, WohT, H_SZ, O_SZ);

  // 2) f_t(bf16) = left * (h_tm1 @ W_fh^T)   M=8192 N=512 K=1024
  gemm_wmma<0><<<dim3(BATCH / TM, F_SZ / TN), 256, 0, stream>>>(
      ftb, F_SZ, nullptr,
      hb, hb, H_SZ, H_SZ,
      Wfhb, Wfhb, H_SZ, H_SZ,
      left, H_SZ);

  // 3) h_t = tanh([f_t|x_t] @ [W_hf|W_hx]^T)   M=8192 N=1024 K=1024 (split 512)
  gemm_wmma<1><<<dim3(BATCH / TM, H_SZ / TN), 256, 0, stream>>>(
      h_out, H_SZ, hb2,
      ftb, xb, F_SZ, F_SZ,
      Whfb, Whxb, F_SZ, F_SZ,
      nullptr, F_SZ + IN_SZ);

  // 4) o_t = h_t @ W_oh + b_o   M=8192 N=1024 K=1024 (WohT = bf16 W_oh^T)
  gemm_wmma<2><<<dim3(BATCH / TM, O_SZ / TN), 256, 0, stream>>>(
      o_out, O_SZ, nullptr,
      hb2, hb2, H_SZ, H_SZ,
      WohT, WohT, H_SZ, H_SZ,
      b_o, H_SZ);
}
